// GRUMemory_5007931867260
// MI455X (gfx1250) — matrix operand order repaired and checked by reading, not compiled, not run on hardware
//
#include <hip/hip_runtime.h>
#include <hip/hip_bf16.h>

// GRU language model: B=512, T=1024, V=27, H=512.
//  - bf16 WMMA (v_wmma_f32_16x16x32_bf16) for all matmuls, fp32 gate math.
//  - One fused per-step GEMM: A = [h_bf16 | x_t(pad 32)] (512 x 544),
//    B = Wcat (2048 x 544), 4 column-groups per 16-wide j-tile (R, Z, HN, IN)
//    so a 64x128 block tile holds every gate piece for the same 32 hidden units.
//  - LDS staging via CDNA5 async global->LDS copies (ASYNCcnt) when the builtins
//    exist, else the proven global_load_b128 -> ds_store_b128 path.
//  - Ping-pong h buffers (fp32 master + bf16 shadow). Decode of h_{t-1} rides in
//    the same launch as step t (8 extra blocks) via WMMA against padded W_dec.
//  - 1 prep kernel + 1025 step launches; the harness graph-captures the stream.

#define Tt   1024
#define Vv   27
#define Hh   512
#define Bb   512
#define KTOT 544      // 512 (h) + 32 (x padded)
#define NKC  17       // K chunks of 32
#define NCOL 2048     // 4 * H output columns

typedef __attribute__((ext_vector_type(16))) __bf16 v16bf;
typedef __attribute__((ext_vector_type(8)))  float  v8f;
typedef int v4i_ __attribute__((ext_vector_type(4)));
typedef __attribute__((address_space(1))) v4i_ as1_v4i;  // global int4
typedef __attribute__((address_space(3))) v4i_ as3_v4i;  // LDS int4

union FragBF { uint4 u[2]; v16bf v; };

#if __has_builtin(__builtin_amdgcn_global_load_async_to_lds_b128) && \
    __has_builtin(__builtin_amdgcn_s_wait_asynccnt)
#define USE_ASYNC 1
#else
#define USE_ASYNC 0
#endif

// 16B global -> LDS copy. Async path: tracked with ASYNCcnt, no VGPR bounce.
__device__ __forceinline__ void cp16(const void* g, void* l) {
#if USE_ASYNC
  __builtin_amdgcn_global_load_async_to_lds_b128(
      (as1_v4i*)(uintptr_t)g, (as3_v4i*)(uintptr_t)l, 0, 0);
#else
  *(uint4*)l = *(const uint4*)g;
#endif
}
__device__ __forceinline__ void cp_wait() {
#if USE_ASYNC
  __builtin_amdgcn_s_wait_asynccnt(0);
#endif
}

__device__ __forceinline__ unsigned short f2bf(float f) {
  unsigned int u = __float_as_uint(f);
  u += 0x7fffu + ((u >> 16) & 1u);          // round-to-nearest-even
  return (unsigned short)(u >> 16);
}
__device__ __forceinline__ float sigm(float x) { return 1.f / (1.f + __expf(-x)); }
__device__ __forceinline__ float tanh_(float x) {
  x = fminf(fmaxf(x, -20.f), 20.f);
  float e = __expf(2.f * x);
  return (e - 1.f) / (e + 1.f);
}

// ---------------------------------------------------------------------------
// Prep: pack Wcat (2048x544 bf16), WdecP (32x512 bf16, rows>=27 zero), zero h0.
// ---------------------------------------------------------------------------
__global__ void gru_prep(const float* __restrict__ W_ih, const float* __restrict__ W_hh,
                         const float* __restrict__ W_dec,
                         unsigned short* __restrict__ Wcat, unsigned short* __restrict__ WdecP,
                         float* __restrict__ hf0, unsigned short* __restrict__ hb0) {
  const int stride = gridDim.x * blockDim.x;
  const int tid0 = blockIdx.x * blockDim.x + threadIdx.x;

  for (int i = tid0; i < NCOL * KTOT; i += stride) {
    int crow = i / KTOT, k = i - crow * KTOT;
    int jt = crow >> 6, grp = (crow >> 4) & 3, jm = crow & 15;
    int j = (jt << 4) | jm;
    float val = 0.f;
    if (grp == 0) {
      val = (k < Hh) ? W_hh[(size_t)j * Hh + k]
                     : ((k - Hh) < Vv ? W_ih[(size_t)j * Vv + (k - Hh)] : 0.f);
    } else if (grp == 1) {
      int g = Hh + j;
      val = (k < Hh) ? W_hh[(size_t)g * Hh + k]
                     : ((k - Hh) < Vv ? W_ih[(size_t)g * Vv + (k - Hh)] : 0.f);
    } else if (grp == 2) {
      int g = 2 * Hh + j;
      val = (k < Hh) ? W_hh[(size_t)g * Hh + k] : 0.f;
    } else {
      int g = 2 * Hh + j;
      val = (k < Hh) ? 0.f : ((k - Hh) < Vv ? W_ih[(size_t)g * Vv + (k - Hh)] : 0.f);
    }
    Wcat[i] = f2bf(val);
  }
  for (int i = tid0; i < 32 * Hh; i += stride) {
    int vrow = i >> 9, k = i & 511;
    WdecP[i] = (vrow < Vv) ? f2bf(W_dec[(size_t)vrow * Hh + k]) : (unsigned short)0;
  }
  for (int i = tid0; i < Bb * Hh; i += stride) { hf0[i] = 0.f; hb0[i] = 0; }
}

// ---------------------------------------------------------------------------
// Step t: blocks [0,128): GEMM (512 x 2048, K=544) + gates + h update.
//         blocks [128,136): decode h_{t-1} -> logits[:, t-1, :].
// ---------------------------------------------------------------------------
__global__ __launch_bounds__(256) void gru_step(
    const float* __restrict__ x, const float* __restrict__ b_ih,
    const float* __restrict__ b_hh, const float* __restrict__ b_dec,
    const unsigned short* __restrict__ Wcat, const unsigned short* __restrict__ WdecP,
    const float* __restrict__ hfp, float* __restrict__ hfc,
    const unsigned short* __restrict__ hbp, unsigned short* __restrict__ hbc,
    float* __restrict__ out, int t) {
  __shared__ __align__(16) char smem[33792];  // staging double-buffer (30720B) / acc (33792B)
  const int tid = threadIdx.x;
  const int lid = tid & 31;
  const int w   = tid >> 5;

  if (blockIdx.x < 128) {
    if (t >= Tt) return;  // final launch is decode-only
    const int bM  = blockIdx.x >> 4;   // 8 row-blocks of 64
    const int bN  = blockIdx.x & 15;   // 16 col-blocks of 128
    const int rb0 = bM * 64;
    const int cb0 = bN * 128;
    char* As[2] = { smem,        smem + 15360 };
    char* Bs[2] = { smem + 5120, smem + 5120 + 15360 };

    const int mb = (w >> 2) * 32;  // wave -> 2x2 tiles of 16x16 within 64x128
    const int nb = (w & 3)  * 32;
    v8f c00 = {}, c01 = {}, c10 = {}, c11 = {};

    auto stageA = [&](int kc, char* dst) {
      int row = tid >> 2, c8 = (tid & 3) * 8;  // 64 rows x 32 k, 16B per thread
      if (kc < 16) {
        cp16(hbp + (size_t)(rb0 + row) * Hh + kc * 32 + c8,
             dst + row * 80 + c8 * 2);
      } else {  // x_t chunk: fp32 -> bf16, pad V 27 -> 32 (VGPR path: needs convert)
        unsigned short tmp[8];
        const float* xr = x + (size_t)(rb0 + row) * (Tt * Vv) + (size_t)t * Vv;
        #pragma unroll
        for (int i = 0; i < 8; ++i) {
          int v = c8 + i;
          tmp[i] = (v < Vv) ? f2bf(xr[v]) : (unsigned short)0;
        }
        *(uint4*)(dst + row * 80 + c8 * 2) = *(const uint4*)tmp;
      }
    };
    auto stageB = [&](int kc, char* dst) {
      int g = tid >> 1, c16 = (tid & 1) * 16;  // 128 rows x 32 k, 32B per thread
      const unsigned short* src = Wcat + (size_t)(cb0 + g) * KTOT + kc * 32 + c16;
      char* d = dst + g * 80 + c16 * 2;
      cp16(src,     d);
      cp16(src + 8, d + 16);
    };

    stageA(0, As[0]); stageB(0, Bs[0]);
    cp_wait();
    __syncthreads();
    for (int kc = 0; kc < NKC; ++kc) {
      const int cur = kc & 1;
      if (kc + 1 < NKC) { stageA(kc + 1, As[cur ^ 1]); stageB(kc + 1, Bs[cur ^ 1]); }

      FragBF a0, a1, b0, b1;
      {
        // A frag (16x32): lane half 0 -> K 0..15, half 1 -> K 16..31 (contiguous, matches B)
        int row = mb + (lid & 15), k0 = (lid >> 4) * 16;
        const char* p0 = As[cur] + row * 80 + k0 * 2;
        a0.u[0] = *(const uint4*)p0;  a0.u[1] = *(const uint4*)(p0 + 16);
        const char* p1 = As[cur] + (row + 16) * 80 + k0 * 2;
        a1.u[0] = *(const uint4*)p1;  a1.u[1] = *(const uint4*)(p1 + 16);
        // B frag (32x16): lanes 0-15 K=0..15, lanes 16-31 K=16..31 (contiguous 32B)
        int g = nb + (lid & 15), kb = (lid >> 4) * 16;
        const char* q0 = Bs[cur] + g * 80 + kb * 2;
        b0.u[0] = *(const uint4*)q0;  b0.u[1] = *(const uint4*)(q0 + 16);
        const char* q1 = Bs[cur] + (g + 16) * 80 + kb * 2;
        b1.u[0] = *(const uint4*)q1;  b1.u[1] = *(const uint4*)(q1 + 16);
      }
      c00 = __builtin_amdgcn_wmma_f32_16x16x32_bf16(false, a0.v, false, b0.v, (short)0, c00, false, false);
      c01 = __builtin_amdgcn_wmma_f32_16x16x32_bf16(false, a0.v, false, b1.v, (short)0, c01, false, false);
      c10 = __builtin_amdgcn_wmma_f32_16x16x32_bf16(false, a1.v, false, b0.v, (short)0, c10, false, false);
      c11 = __builtin_amdgcn_wmma_f32_16x16x32_bf16(false, a1.v, false, b1.v, (short)0, c11, false, false);
      cp_wait();
      __syncthreads();
    }

    // Spill accumulators to LDS (aliases staging buffers; all LDS reads done).
    float* accS = (float*)smem;  // 64 x 132 fp32
    {
      int rbase = (lid >> 4) * 8, col = lid & 15;
      #pragma unroll
      for (int v = 0; v < 8; ++v) {
        accS[(mb + rbase + v) * 132 + nb + col]           = c00[v];
        accS[(mb + rbase + v) * 132 + nb + 16 + col]      = c01[v];
        accS[(mb + 16 + rbase + v) * 132 + nb + col]      = c10[v];
        accS[(mb + 16 + rbase + v) * 132 + nb + 16 + col] = c11[v];
      }
    }
    __syncthreads();

    // Gates + hidden-state update: 64 rows x 32 hidden units per block.
    {
      int r = tid >> 2, j0 = (tid & 3) * 8;
      int bg = rb0 + r;
      #pragma unroll
      for (int jj = 0; jj < 8; ++jj) {
        int j = j0 + jj;
        int jt = j >> 4, jm = j & 15;
        const float* arow = accS + r * 132 + jt * 64;
        float aR  = arow[jm];
        float aZ  = arow[16 + jm];
        float aHN = arow[32 + jm];
        float aIN = arow[48 + jm];
        int jg = bN * 32 + j;
        float rg = sigm(aR + b_ih[jg] + b_hh[jg]);
        float zg = sigm(aZ + b_ih[Hh + jg] + b_hh[Hh + jg]);
        float ng = tanh_(aIN + b_ih[2 * Hh + jg] + rg * (aHN + b_hh[2 * Hh + jg]));
        float ho = hfp[(size_t)bg * Hh + jg];
        float hn = (1.f - zg) * ng + zg * ho;
        hfc[(size_t)bg * Hh + jg] = hn;
        hbc[(size_t)bg * Hh + jg] = f2bf(hn);
      }
    }
  } else {
    // ---- decode h_{t-1} (the buffer step t reads) into logits[:, t-1, :] ----
    if (t == 0) return;
    const int db  = blockIdx.x - 128;
    const int rb0 = db * 64;
    char* Ad[2] = { smem, smem + 5120 };   // double-buffered 64x32 bf16 tiles
    const int mb = (w >> 1) * 16;  // 8 waves -> 4(M) x 2(N) tiles
    const int nb = (w & 1)  * 16;
    v8f c = {};

    auto stageD = [&](int kc, char* dst) {
      int row = tid >> 2, c8 = (tid & 3) * 8;
      cp16(hbp + (size_t)(rb0 + row) * Hh + kc * 32 + c8,
           dst + row * 80 + c8 * 2);
    };

    stageD(0, Ad[0]);
    cp_wait();
    __syncthreads();
    for (int kc = 0; kc < 16; ++kc) {
      const int cur = kc & 1;
      if (kc + 1 < 16) stageD(kc + 1, Ad[cur ^ 1]);

      FragBF a, b;
      {
        int row = mb + (lid & 15), k0 = (lid >> 4) * 16;
        const char* p0 = Ad[cur] + row * 80 + k0 * 2;
        a.u[0] = *(const uint4*)p0;  a.u[1] = *(const uint4*)(p0 + 16);
        int g = nb + (lid & 15), kb = (lid >> 4) * 16;
        const uint4* q = (const uint4*)(WdecP + (size_t)g * Hh + kc * 32 + kb);
        b.u[0] = q[0];  b.u[1] = q[1];
      }
      c = __builtin_amdgcn_wmma_f32_16x16x32_bf16(false, a.v, false, b.v, (short)0, c, false, false);
      cp_wait();
      __syncthreads();
    }
    {
      int rbase = (lid >> 4) * 8, col = nb + (lid & 15);
      if (col < Vv) {
        float bd = b_dec[col];
        #pragma unroll
        for (int v = 0; v < 8; ++v) {
          int bg = rb0 + mb + rbase + v;
          out[(size_t)bg * (Tt * Vv) + (size_t)(t - 1) * Vv + col] = c[v] + bd;
        }
      }
    }
  }
}

// ---------------------------------------------------------------------------
extern "C" void kernel_launch(void* const* d_in, const int* in_sizes, int n_in,
                              void* d_out, int out_size, void* d_ws, size_t ws_size,
                              hipStream_t stream) {
  const float* x     = (const float*)d_in[0];
  const float* W_ih  = (const float*)d_in[1];
  const float* W_hh  = (const float*)d_in[2];
  const float* b_ih  = (const float*)d_in[3];
  const float* b_hh  = (const float*)d_in[4];
  const float* W_dec = (const float*)d_in[5];
  const float* b_dec = (const float*)d_in[6];
  float* out = (float*)d_out;

  char* p = (char*)d_ws;
  unsigned short* Wcat  = (unsigned short*)p; p += (size_t)NCOL * KTOT * 2;  // 2.2 MB
  unsigned short* WdecP = (unsigned short*)p; p += (size_t)32 * Hh * 2;      // 32 KB
  float* hf0 = (float*)p;          p += (size_t)Bb * Hh * 4;                 // 1 MB
  float* hf1 = (float*)p;          p += (size_t)Bb * Hh * 4;                 // 1 MB
  unsigned short* hb0 = (unsigned short*)p; p += (size_t)Bb * Hh * 2;        // 0.5 MB
  unsigned short* hb1 = (unsigned short*)p; p += (size_t)Bb * Hh * 2;        // 0.5 MB

  gru_prep<<<512, 256, 0, stream>>>(W_ih, W_hh, W_dec, Wcat, WdecP, hf0, hb0);

  for (int t = 0; t <= Tt; ++t) {  // t == Tt: decode-only tail for h_{T-1}
    const float*          hfp = (t & 1) ? hf1 : hf0;
    float*                hfc = (t & 1) ? hf0 : hf1;
    const unsigned short* hbp = (t & 1) ? hb1 : hb0;
    unsigned short*       hbc = (t & 1) ? hb0 : hb1;
    gru_step<<<136, 256, 0, stream>>>(x, b_ih, b_hh, b_dec, Wcat, WdecP,
                                      hfp, hfc, hbp, hbc, out, t);
  }
}
